// VqVae_16406775070748
// MI455X (gfx1250) — matrix operand order repaired and checked by reading, not compiled, not run on hardware
//
#include <hip/hip_runtime.h>
#include <hip/hip_bf16.h>
#include <math.h>

typedef __attribute__((ext_vector_type(16))) _Float16 v16h;
typedef __attribute__((ext_vector_type(8)))  _Float16 v8h;
typedef __attribute__((ext_vector_type(8)))  float    v8f;

// ---------------------------------------------------------------------------
// Weight preprocessing into a ZERO-PADDED f16 arena:
//   dst is (MP x KP) row-major, MP = ceil(COUT/16)*16, KP = ceil(KDIM/32)*32.
//   TRANS=false: conv OIHW  src[oc*KDIM + k]
//   TRANS=true : tconv (in,out,khw) src[(ic*COUT+oc)*KHW + r] -> row-major
// Padding rows/cols are zero, so the GEMM A-path needs NO bounds checks.
// ---------------------------------------------------------------------------
template<bool TRANS>
__global__ __launch_bounds__(256) void wprep_kernel(
    const float* __restrict__ s, _Float16* __restrict__ d,
    int COUT, int KDIM, int KP, int MP, int KHW) {
  const int i = blockIdx.x * 256 + threadIdx.x;
  if (i >= MP * KP) return;
  const int oc = i / KP;
  const int k  = i - oc * KP;
  float v = 0.0f;
  if (oc < COUT && k < KDIM) {
    if (TRANS) {
      const int ic = k / KHW;
      const int r  = k - ic * KHW;
      v = s[(ic * COUT + oc) * KHW + r];
    } else {
      v = s[oc * KDIM + k];
    }
  }
  d[i] = (_Float16)v;
}

// ---------------------------------------------------------------------------
// Unified implicit-GEMM conv / conv-transpose with WMMA f32_16x16x32_f16.
//   GEMM: M = COUT, N = B*OH*OW, K = CIN*KH*KW
//   Block = 256 threads = 8 waves = MW (M-tiles) x NWAVE (N-groups).
//   Waves in an N-group share ONE LDS-staged 32(K)x16(N) f16 B tile, staged
//   branchlessly (clamped addresses + cndmask select) with two K values
//   packed per 32-bit ds_store.  A fragment: two unconditional aligned 16B
//   vector loads from the padded f16 weight arena.  Epilogue uses one base
//   pointer + compile-time immediate offsets; M-guard folds away when
//   COUT % 16 == 0.
// Fragment layouts used by this kernel:
//   A (16x32): M = lane&15 ; K = k0 + (lane>>4)*16 + e
//   B (32x16): N = lane&15 ; K = k0 + (lane>>4)*16 + e
//   C/D:       M = r + 8*(lane>>4) ; N = lane&15
// ---------------------------------------------------------------------------
template<int CIN,int COUT,int KH,int KW,int STR,int PAD,int H,int W,
         int MW,int NWAVE, typename TIN, typename TOUT,
         bool TCONV, bool IN_RELU, int OUT_ACT, bool HAS_RES, bool HAS_BIAS>
__global__ __launch_bounds__(256) void gemm_conv(
    const TIN* __restrict__ x, const _Float16* __restrict__ wgt,
    const float* __restrict__ bias, const TIN* __restrict__ res,
    TOUT* __restrict__ out, int NB)
{
  constexpr int  OH    = TCONV ? ((H-1)*STR - 2*PAD + KH) : ((H + 2*PAD - KH)/STR + 1);
  constexpr int  OW    = TCONV ? ((W-1)*STR - 2*PAD + KW) : ((W + 2*PAD - KW)/STR + 1);
  constexpr int  KDIM  = CIN*KH*KW;
  constexpr int  KP    = ((KDIM + 31)/32)*32;   // padded K extent of weight arena
  constexpr int  RSW   = 20;                    // LDS row stride in 32-bit words (=40 halves)
  constexpr int  ITER  = 256/(MW*32);           // packed words staged per thread
  constexpr bool MFULL = (COUT % 16) == 0;      // M-guard statically removable

  __shared__ alignas(16) unsigned btw[NWAVE][16*RSW];

  const int Ntot = NB*OH*OW;
  const int w    = threadIdx.x >> 5;
  const int lane = threadIdx.x & 31;
  const int mw   = w % MW;                      // M tile within block
  const int ng   = w / MW;                      // N group within block
  const int tg   = threadIdx.x - ng*(MW*32);    // thread id within N group
  const int half = lane >> 4;
  const int l15  = lane & 15;

  const int ntile  = (blockIdx.x*NWAVE + ng)*16;
  const int octile = (blockIdx.y*MW + mw)*16;
  const _Float16* warow = wgt + (octile + l15)*KP;   // padded: always valid

  // branchless im2col gather of one (n, k) element
  auto gather = [&](int n, int k) -> _Float16 {
    const int nc = (n < Ntot) ? n : 0;          // clamped: address always valid
    const int kc = (k < KDIM) ? k : 0;
    const int ow = nc % OW;
    const int oh = (nc/OW) % OH;
    const int bb = nc/(OW*OH);
    const int ic = kc / (KH*KW);
    const int r  = kc - ic*(KH*KW);
    const int kh = r / KW;
    const int kw = r - kh*KW;
    bool ok = (n < Ntot) && (k < KDIM);
    int ih, iw;
    if (!TCONV) {
      ih = oh*STR - PAD + kh;
      iw = ow*STR - PAD + kw;
      ok = ok && (ih >= 0) && (ih < H) && (iw >= 0) && (iw < W);
    } else {
      const int th = oh + PAD - kh;
      const int tw = ow + PAD - kw;
      ih = th / STR;
      iw = tw / STR;
      ok = ok && (th >= 0) && (tw >= 0) && ((th % STR) == 0) && ((tw % STR) == 0)
              && (ih < H) && (iw < W);
    }
    const int ihc = ih < 0 ? 0 : (ih >= H ? H-1 : ih);
    const int iwc = iw < 0 ? 0 : (iw >= W ? W-1 : iw);
    float v = (float)x[((bb*CIN + ic)*H + ihc)*W + iwc];   // unconditional load
    if (IN_RELU) v = fmaxf(v, 0.0f);
    v = ok ? v : 0.0f;                                     // predicate select
    return (_Float16)v;
  };

  v8f acc = {};
  for (int k0 = 0; k0 < KP; k0 += 32) {
    __syncthreads();
    // ---- cooperative staging: 2 adjacent K values packed per ds_store_b32
#pragma unroll
    for (int j = 0; j < ITER; ++j) {
      const int W0 = tg + j*(MW*32);
      const int nn = W0 & 15;                   // N within tile
      const int k2 = W0 >> 4;                   // word column (2 K values)
      const _Float16 h0 = gather(ntile + nn, k0 + 2*k2);
      const _Float16 h1 = gather(ntile + nn, k0 + 2*k2 + 1);
      const unsigned u = ((unsigned)__builtin_bit_cast(unsigned short, h1) << 16)
                       |  (unsigned)__builtin_bit_cast(unsigned short, h0);
      btw[ng][nn*RSW + k2] = u;
    }
    __syncthreads();

    // ---- A fragment: two unconditional aligned 16B vector loads
    v16h af;
    {
      const v8h* pa = (const v8h*)(warow + k0 + half*16);
      const v8h lo = pa[0];
      const v8h hi = pa[1];                     // +8 halves
#pragma unroll
      for (int e = 0; e < 8; ++e) { af[e] = lo[e]; af[e+8] = hi[e]; }
    }

    // ---- B fragment: 32 contiguous bytes from LDS
    v16h bf;
    {
      const _Float16* pb = (const _Float16*)&btw[ng][0] + l15*(2*RSW) + half*16;
#pragma unroll
      for (int e = 0; e < 16; ++e) bf[e] = pb[e];
    }

    acc = __builtin_amdgcn_wmma_f32_16x16x32_f16(false, af, false, bf,
                                                 (short)0, acc, false, false);
  }

  // ---- fused epilogue: base pointer + immediate offsets, static M-guard
  const int nS = ntile + l15;
  if (nS < Ntot) {
    const int ow  = nS % OW;
    const int oh  = (nS/OW) % OH;
    const int bb  = nS/(OW*OH);
    const int ocb = octile + 8*half;            // first oc of this lane's column
    const int base = ((bb*COUT + ocb)*OH + oh)*OW + ow;
    TOUT* po = out + base;
    const TIN* pr = HAS_RES ? (res + base) : (const TIN*)nullptr;
#pragma unroll
    for (int r8 = 0; r8 < 8; ++r8) {
      if (MFULL || (ocb + r8) < COUT) {
        float v = acc[r8];
        if (HAS_BIAS) v += bias[ocb + r8];
        if (HAS_RES)  v += (float)pr[r8*(OH*OW)];
        if (OUT_ACT == 1) v = fmaxf(v, 0.0f);
        if (OUT_ACT == 2) v = tanhf(v);
        po[r8*(OH*OW)] = (TOUT)v;
      }
    }
  }
}

// ---------------------------------------------------------------------------
// Vector quantizer: 512x8 codebook in LDS (16 KB), per-thread 512-way argmin
// (first-min tie-break, like jnp.argmin), emits gathered rows as f16 NCHW.
// ---------------------------------------------------------------------------
__global__ __launch_bounds__(256) void vq_kernel(
    const float* __restrict__ ze, const float* __restrict__ cb,
    _Float16* __restrict__ zq, int Ntot)
{
  __shared__ float cbs[512*8];
  for (int i = threadIdx.x; i < 512*8; i += 256) cbs[i] = cb[i];
  __syncthreads();

  const int n = blockIdx.x*256 + threadIdx.x;
  if (n >= Ntot) return;
  const int gw = n & 63;
  const int gh = (n >> 6) & 63;
  const int b  = n >> 12;

  float z[8];
#pragma unroll
  for (int d = 0; d < 8; ++d) z[d] = ze[((b*8 + d)*64 + gh)*64 + gw];

  float best = 3.4e38f;
  int   bi   = 0;
  for (int k = 0; k < 512; ++k) {
    float s = 0.0f;
#pragma unroll
    for (int d = 0; d < 8; ++d) {
      const float df = z[d] - cbs[k*8 + d];
      s = fmaf(df, df, s);
    }
    if (s < best) { best = s; bi = k; }
  }
#pragma unroll
  for (int d = 0; d < 8; ++d)
    zq[((b*8 + d)*64 + gh)*64 + gw] = (_Float16)cbs[bi*8 + d];
}

// ---------------------------------------------------------------------------
// Host driver
// ---------------------------------------------------------------------------
extern "C" void kernel_launch(void* const* d_in, const int* in_sizes, int n_in,
                              void* d_out, int out_size, void* d_ws, size_t ws_size,
                              hipStream_t stream) {
  (void)in_sizes; (void)n_in; (void)out_size; (void)ws_size;

  const float* x        = (const float*)d_in[0];
  const float* enc_w1   = (const float*)d_in[1];
  const float* enc_b1   = (const float*)d_in[2];
  const float* enc_w2   = (const float*)d_in[3];
  const float* enc_b2   = (const float*)d_in[4];
  const float* enc_w3   = (const float*)d_in[5];
  const float* enc_b3   = (const float*)d_in[6];
  const float* er_w3[3] = { (const float*)d_in[7],  (const float*)d_in[9],  (const float*)d_in[11] };
  const float* er_w1[3] = { (const float*)d_in[8],  (const float*)d_in[10], (const float*)d_in[12] };
  const float* pre_w    = (const float*)d_in[13];
  const float* pre_b    = (const float*)d_in[14];
  const float* codebook = (const float*)d_in[15];
  const float* dec_w1   = (const float*)d_in[16];
  const float* dec_b1   = (const float*)d_in[17];
  const float* dr_w3[3] = { (const float*)d_in[18], (const float*)d_in[20], (const float*)d_in[22] };
  const float* dr_w1[3] = { (const float*)d_in[19], (const float*)d_in[21], (const float*)d_in[23] };
  const float* dec_tw1  = (const float*)d_in[24];
  const float* dec_tb1  = (const float*)d_in[25];
  const float* dec_tw2  = (const float*)d_in[26];
  const float* dec_tb2  = (const float*)d_in[27];
  float* recon = (float*)d_out;

  const int NB = 8;

  // -------- workspace layout --------
  char* ws = (char*)d_ws;
  _Float16* a1 = (_Float16*)(ws);               // 8*64*128*128 f16 = 16 MiB (reused after encoder)
  _Float16* a2 = (_Float16*)(ws + (16u << 20)); // 8 MiB
  _Float16* a3 = (_Float16*)(ws + (24u << 20)); // 8 MiB
  _Float16* t  = (_Float16*)(ws + (32u << 20)); // 4 MiB
  float*    ze = (float*)   (ws + (36u << 20)); // 1 MiB
  _Float16* zq = (_Float16*)(ws + (37u << 20)); // 0.5 MiB

  size_t wo = (size_t)38u << 20;                // padded f16 weight arena
  auto alloc16 = [&](size_t nelem) {
    _Float16* p = (_Float16*)(ws + wo);
    wo += ((nelem*2 + 255)/256)*256;
    return p;
  };
  // padded sizes: MP x KP
  _Float16* q_enc_w1 = alloc16(64*64);          // COUT=64 KDIM=48 ->KP=64
  _Float16* q_enc_w2 = alloc16(128*1024);
  _Float16* q_enc_w3 = alloc16(128*1152);
  _Float16* q_er_w3[3] = { alloc16(64*1152), alloc16(64*1152), alloc16(64*1152) };
  _Float16* q_er_w1[3] = { alloc16(128*64),  alloc16(128*64),  alloc16(128*64)  };
  _Float16* q_pre_w  = alloc16(16*128);         // COUT=8 -> MP=16
  _Float16* q_dec_w1 = alloc16(128*96);         // KDIM=72 -> KP=96
  _Float16* q_dr_w3[3] = { alloc16(64*1152), alloc16(64*1152), alloc16(64*1152) };
  _Float16* q_dr_w1[3] = { alloc16(128*64),  alloc16(128*64),  alloc16(128*64)  };
  _Float16* q_tw1 = alloc16(64*2048);           // tconv repack (out, in*16)
  _Float16* q_tw2 = alloc16(16*1024);           // COUT=3 -> MP=16

  // -------- weight preprocessing (padded arena) --------
  auto prep = [&](const float* s, _Float16* d, int cout, int kdim, int khw, bool trans) {
    const int kp = ((kdim + 31)/32)*32;
    const int mp = ((cout + 15)/16)*16;
    const dim3 g((unsigned)((mp*kp + 255)/256)), b(256);
    if (trans) wprep_kernel<true> <<<g, b, 0, stream>>>(s, d, cout, kdim, kp, mp, khw);
    else       wprep_kernel<false><<<g, b, 0, stream>>>(s, d, cout, kdim, kp, mp, khw);
  };
  prep(enc_w1, q_enc_w1, 64, 48, 16, false);
  prep(enc_w2, q_enc_w2, 128, 1024, 16, false);
  prep(enc_w3, q_enc_w3, 128, 1152, 9, false);
  for (int i = 0; i < 3; ++i) {
    prep(er_w3[i], q_er_w3[i], 64, 1152, 9, false);
    prep(er_w1[i], q_er_w1[i], 128, 64, 1, false);
    prep(dr_w3[i], q_dr_w3[i], 64, 1152, 9, false);
    prep(dr_w1[i], q_dr_w1[i], 128, 64, 1, false);
  }
  prep(pre_w,  q_pre_w,  8, 128, 1, false);
  prep(dec_w1, q_dec_w1, 128, 72, 9, false);
  prep(dec_tw1, q_tw1, 64, 2048, 16, true);
  prep(dec_tw2, q_tw2, 3, 1024, 16, true);

  const dim3 blk(256);
  const int N128 = NB*128*128;   // 131072
  const int N64  = NB*64*64;     // 32768
  const int N256 = NB*256*256;   // 524288
  auto grid = [](int ntot, int cout, int mwv, int nwv) {
    return dim3((unsigned)((ntot + 16*nwv - 1)/(16*nwv)),
                (unsigned)((cout + 16*mwv - 1)/(16*mwv)));
  };

  // ---------------- Encoder ----------------
  gemm_conv<3,64,4,4,2,1,256,256, 4,2, float,_Float16, false,false,1,false,true>
      <<<grid(N128,64,4,2), blk, 0, stream>>>(x, q_enc_w1, enc_b1, (const float*)nullptr, a1, NB);
  gemm_conv<64,128,4,4,2,1,128,128, 8,1, _Float16,_Float16, false,false,1,false,true>
      <<<grid(N64,128,8,1), blk, 0, stream>>>(a1, q_enc_w2, enc_b2, (const _Float16*)nullptr, a2, NB);
  gemm_conv<128,128,3,3,1,1,64,64, 8,1, _Float16,_Float16, false,false,1,false,true>
      <<<grid(N64,128,8,1), blk, 0, stream>>>(a2, q_enc_w3, enc_b3, (const _Float16*)nullptr, a3, NB);

  // enc residual stack: x += conv1x1(relu(conv3x3(relu(x))))
  gemm_conv<128,64,3,3,1,1,64,64, 4,2, _Float16,_Float16, false,true,1,false,false>
      <<<grid(N64,64,4,2), blk, 0, stream>>>(a3, q_er_w3[0], nullptr, (const _Float16*)nullptr, t, NB);
  gemm_conv<64,128,1,1,1,0,64,64, 8,1, _Float16,_Float16, false,false,0,true,false>
      <<<grid(N64,128,8,1), blk, 0, stream>>>(t, q_er_w1[0], nullptr, a3, a2, NB);
  gemm_conv<128,64,3,3,1,1,64,64, 4,2, _Float16,_Float16, false,true,1,false,false>
      <<<grid(N64,64,4,2), blk, 0, stream>>>(a2, q_er_w3[1], nullptr, (const _Float16*)nullptr, t, NB);
  gemm_conv<64,128,1,1,1,0,64,64, 8,1, _Float16,_Float16, false,false,0,true,false>
      <<<grid(N64,128,8,1), blk, 0, stream>>>(t, q_er_w1[1], nullptr, a2, a3, NB);
  gemm_conv<128,64,3,3,1,1,64,64, 4,2, _Float16,_Float16, false,true,1,false,false>
      <<<grid(N64,64,4,2), blk, 0, stream>>>(a3, q_er_w3[2], nullptr, (const _Float16*)nullptr, t, NB);
  gemm_conv<64,128,1,1,1,0,64,64, 8,1, _Float16,_Float16, false,false,0,true,false>
      <<<grid(N64,128,8,1), blk, 0, stream>>>(t, q_er_w1[2], nullptr, a3, a2, NB);

  // pre-VQ projection (fuses trailing ReLU of res-stack via IN_RELU)
  gemm_conv<128,8,1,1,1,0,64,64, 1,8, _Float16,float, false,true,0,false,true>
      <<<grid(N64,8,1,8), blk, 0, stream>>>(a2, q_pre_w, pre_b, (const _Float16*)nullptr, ze, NB);

  // ---------------- Vector quantizer ----------------
  vq_kernel<<<dim3((unsigned)((N64 + 255)/256)), dim3(256), 0, stream>>>(ze, codebook, zq, N64);

  // ---------------- Decoder ----------------
  gemm_conv<8,128,3,3,1,1,64,64, 8,1, _Float16,_Float16, false,false,1,false,true>
      <<<grid(N64,128,8,1), blk, 0, stream>>>(zq, q_dec_w1, dec_b1, (const _Float16*)nullptr, a2, NB);

  gemm_conv<128,64,3,3,1,1,64,64, 4,2, _Float16,_Float16, false,true,1,false,false>
      <<<grid(N64,64,4,2), blk, 0, stream>>>(a2, q_dr_w3[0], nullptr, (const _Float16*)nullptr, t, NB);
  gemm_conv<64,128,1,1,1,0,64,64, 8,1, _Float16,_Float16, false,false,0,true,false>
      <<<grid(N64,128,8,1), blk, 0, stream>>>(t, q_dr_w1[0], nullptr, a2, a3, NB);
  gemm_conv<128,64,3,3,1,1,64,64, 4,2, _Float16,_Float16, false,true,1,false,false>
      <<<grid(N64,64,4,2), blk, 0, stream>>>(a3, q_dr_w3[1], nullptr, (const _Float16*)nullptr, t, NB);
  gemm_conv<64,128,1,1,1,0,64,64, 8,1, _Float16,_Float16, false,false,0,true,false>
      <<<grid(N64,128,8,1), blk, 0, stream>>>(t, q_dr_w1[1], nullptr, a3, a2, NB);
  gemm_conv<128,64,3,3,1,1,64,64, 4,2, _Float16,_Float16, false,true,1,false,false>
      <<<grid(N64,64,4,2), blk, 0, stream>>>(a2, q_dr_w3[2], nullptr, (const _Float16*)nullptr, t, NB);
  gemm_conv<64,128,1,1,1,0,64,64, 8,1, _Float16,_Float16, false,false,0,true,false>
      <<<grid(N64,128,8,1), blk, 0, stream>>>(t, q_dr_w1[2], nullptr, a2, a3, NB);

  // tconv1: fuses trailing decoder-res ReLU (IN_RELU) + output ReLU
  gemm_conv<128,64,4,4,2,1,64,64, 4,2, _Float16,_Float16, true,true,1,false,true>
      <<<grid(N128,64,4,2), blk, 0, stream>>>(a3, q_tw1, dec_tb1, (const _Float16*)nullptr, a1, NB);
  // tconv2 -> tanh -> f32 output
  gemm_conv<64,3,4,4,2,1,128,128, 1,8, _Float16,float, true,false,2,false,true>
      <<<grid(N256,3,1,8), blk, 0, stream>>>(a1, q_tw2, dec_tb2, (const _Float16*)nullptr, recon, NB);
}
